// MagNet_23630910063296
// MI455X (gfx1250) — matrix operand order repaired and checked by reading, not compiled, not run on hardware
//
#include <hip/hip_runtime.h>
#include <math.h>

typedef __attribute__((ext_vector_type(16))) _Float16 v16h;
typedef __attribute__((ext_vector_type(8)))  _Float16 v8h;
typedef __attribute__((ext_vector_type(8)))  float    v8f;
typedef __attribute__((ext_vector_type(4)))  float    v4f;

// ---------------------------------------------------------------------------
// Fused dual-chain WMMA GEMM + complex-ReLU epilogue.
//   accA = Aa[nrows,kdim] @ W[kdim,64]
//   accB = Ab[nrows,kdim] @ W[kdim,64]
//   r = accA - accB + bias ; i = accA + accB + bias ; mask = (r >= 0)
//   outR = mask*r ; outI = mask*i
// f32 in/out, f16 multiply via v_wmma_f32_16x16x32_f16, f32 accumulate.
// One wave = 16 rows x 64 cols (4+4 accumulators). kdim multiple of 32, <=384.
// B fragments are read via persistent LDS pointers advanced by a constant so
// address updates never clobber WMMA source VGPRs (avoids WAR v_nop hazard).
// ---------------------------------------------------------------------------
__global__ __launch_bounds__(256)
void gemm2_wmma_relu(const float* __restrict__ Aa, const float* __restrict__ Ab,
                     const float* __restrict__ W,     // [kdim, 64] row-major
                     const float* __restrict__ bias,  // [64]
                     float* __restrict__ outR, float* __restrict__ outI,
                     int nrows, int kdim)
{
    __shared__ _Float16 Wt[64 * 384];   // W transposed: Wt[n*kdim + k], max 48 KB

    const int tid = threadIdx.x;
    for (int idx = tid; idx < kdim * 64; idx += 256) {
        const int k = idx >> 6;         // W row-major [kdim, 64]
        const int n = idx & 63;
        Wt[n * kdim + k] = (_Float16)W[idx];
    }
    __syncthreads();

    const int wave = tid >> 5;
    const int lane = tid & 31;
    const int lo   = lane & 15;
    const int hi   = lane >> 4;
    const int row0 = (blockIdx.x * 8 + wave) * 16;
    if (row0 >= nrows) return;          // full 16-row tiles only (N % 16 == 0)

    v8f accA[4], accB[4];
    #pragma unroll
    for (int cb = 0; cb < 4; ++cb)
        #pragma unroll
        for (int r = 0; r < 8; ++r) { accA[cb][r] = 0.0f; accB[cb][r] = 0.0f; }

    // persistent streaming pointers
    const float* arA = Aa + (size_t)(row0 + lo) * kdim + hi * 8;
    const float* arB = Ab + (size_t)(row0 + lo) * kdim + hi * 8;
    const _Float16* wp0 = Wt + (size_t)(0 * 16 + lo) * kdim + hi * 8;
    const _Float16* wp1 = Wt + (size_t)(1 * 16 + lo) * kdim + hi * 8;
    const _Float16* wp2 = Wt + (size_t)(2 * 16 + lo) * kdim + hi * 8;
    const _Float16* wp3 = Wt + (size_t)(3 * 16 + lo) * kdim + hi * 8;

    for (int k0 = 0; k0 < kdim; k0 += 32) {
        __builtin_prefetch(arA + 64, 0, 0);     // global_prefetch_b8
        __builtin_prefetch(arB + 64, 0, 0);

        // ---- issue all loads for this K-step ----
        v4f a0 = *(const v4f*)(arA);
        v4f a1 = *(const v4f*)(arA + 4);
        v4f a2 = *(const v4f*)(arA + 16);
        v4f a3 = *(const v4f*)(arA + 20);
        v4f c0 = *(const v4f*)(arB);
        v4f c1 = *(const v4f*)(arB + 4);
        v4f c2 = *(const v4f*)(arB + 16);
        v4f c3 = *(const v4f*)(arB + 20);
        v8h w00 = *(const v8h*)(wp0);  v8h w01 = *(const v8h*)(wp0 + 16);
        v8h w10 = *(const v8h*)(wp1);  v8h w11 = *(const v8h*)(wp1 + 16);
        v8h w20 = *(const v8h*)(wp2);  v8h w21 = *(const v8h*)(wp2 + 16);
        v8h w30 = *(const v8h*)(wp3);  v8h w31 = *(const v8h*)(wp3 + 16);
        arA += 32; arB += 32;
        wp0 += 32; wp1 += 32; wp2 += 32; wp3 += 32;

        // ---- build fragments ----
        // A layout: lane<16 holds K {0..7, 16..23}+k0 of its row; lane>=16: {8..15, 24..31}+k0
        v16h aA, aB;
        #pragma unroll
        for (int j = 0; j < 4; ++j) {
            aA[j]      = (_Float16)a0[j];
            aA[4 + j]  = (_Float16)a1[j];
            aA[8 + j]  = (_Float16)a2[j];
            aA[12 + j] = (_Float16)a3[j];
            aB[j]      = (_Float16)c0[j];
            aB[4 + j]  = (_Float16)c1[j];
            aB[8 + j]  = (_Float16)c2[j];
            aB[12 + j] = (_Float16)c3[j];
        }
        // B layout: lane<16 -> K {0..7, 16..23}+k0 of its column; lane>=16 -> K {8..15, 24..31}+k0
        v16h b0, b1, b2, b3;
        #pragma unroll
        for (int j = 0; j < 8; ++j) {
            b0[j] = w00[j]; b0[8 + j] = w01[j];
            b1[j] = w10[j]; b1[8 + j] = w11[j];
            b2[j] = w20[j]; b2[8 + j] = w21[j];
            b3[j] = w30[j]; b3[8 + j] = w31[j];
        }

        // ---- 8-WMMA burst (B shared by both chains) ----
        accA[0] = __builtin_amdgcn_wmma_f32_16x16x32_f16(false, aA, false, b0, (short)0, accA[0], false, false);
        accB[0] = __builtin_amdgcn_wmma_f32_16x16x32_f16(false, aB, false, b0, (short)0, accB[0], false, false);
        accA[1] = __builtin_amdgcn_wmma_f32_16x16x32_f16(false, aA, false, b1, (short)0, accA[1], false, false);
        accB[1] = __builtin_amdgcn_wmma_f32_16x16x32_f16(false, aB, false, b1, (short)0, accB[1], false, false);
        accA[2] = __builtin_amdgcn_wmma_f32_16x16x32_f16(false, aA, false, b2, (short)0, accA[2], false, false);
        accB[2] = __builtin_amdgcn_wmma_f32_16x16x32_f16(false, aB, false, b2, (short)0, accB[2], false, false);
        accA[3] = __builtin_amdgcn_wmma_f32_16x16x32_f16(false, aA, false, b3, (short)0, accA[3], false, false);
        accB[3] = __builtin_amdgcn_wmma_f32_16x16x32_f16(false, aB, false, b3, (short)0, accB[3], false, false);
    }

    // epilogue: complex combine + bias + complex ReLU
    #pragma unroll
    for (int cb = 0; cb < 4; ++cb) {
        const float bb = bias[cb * 16 + lo];
        #pragma unroll
        for (int r = 0; r < 8; ++r) {
            const size_t idx = (size_t)(row0 + r + 8 * hi) * 64 + cb * 16 + lo;
            const float a = accA[cb][r];
            const float b = accB[cb][r];
            const float rr = a - b + bb;
            const float ii = a + b + bb;
            const float m = rr >= 0.0f ? 1.0f : 0.0f;
            outR[idx] = m * rr;
            outI[idx] = m * ii;
        }
    }
}

// ---------------------------------------------------------------------------
// Graph / elementwise kernels
// ---------------------------------------------------------------------------
__global__ void k_fill(float* __restrict__ p, float v, int n)
{
    int i = blockIdx.x * blockDim.x + threadIdx.x;
    if (i < n) p[i] = v;
}

__global__ void k_deg(const int* __restrict__ src, const int* __restrict__ dst,
                      float* __restrict__ deg, int E)
{
    int e = blockIdx.x * blockDim.x + threadIdx.x;
    if (e >= E) return;
    atomicAdd(deg + src[e], 0.5f);
    atomicAdd(deg + dst[e], 0.5f);
}

__global__ void k_dinv(const float* __restrict__ deg, float* __restrict__ dinv, int n)
{
    int i = blockIdx.x * blockDim.x + threadIdx.x;
    if (i >= n) return;
    float d = deg[i];
    dinv[i] = d > 0.0f ? rsqrtf(d) : 0.0f;
}

__global__ void k_anorm(const int* __restrict__ src, const int* __restrict__ dst,
                        const float* __restrict__ dinv, float* __restrict__ an, int E)
{
    int e = blockIdx.x * blockDim.x + threadIdx.x;
    if (e >= E) return;
    an[e] = dinv[src[e]] * 0.5f * dinv[dst[e]];
}

// out[n*ldo+f] = alpha*x[n*ldx+f] + beta*y[n*ldy+f]   over n in [0,N), f in [0,F)
__global__ void k_axpby(float* __restrict__ out, int ldo,
                        const float* __restrict__ x, int ldx, float alpha,
                        const float* __restrict__ y, int ldy, float beta,
                        int N, int F)
{
    int idx = blockIdx.x * blockDim.x + threadIdx.x;
    if (idx >= N * F) return;
    const int n = idx / F;
    const int f = idx - n * F;
    out[(size_t)n * ldo + f] = alpha * x[(size_t)n * ldx + f] + beta * y[(size_t)n * ldy + f];
}

// out[dst] += wf*anorm*x[src]; out[src] += wb*anorm*x[dst]   (per 4-feature chunk)
__global__ void k_scatter(const int* __restrict__ src, const int* __restrict__ dst,
                          const float* __restrict__ an,
                          const float* __restrict__ x, int ldx,
                          float* __restrict__ out, int ldo,
                          int E, int F, float wf, float wb)
{
    const int chunks = F >> 2;
    const int total  = E * chunks;
    int idx = blockIdx.x * blockDim.x + threadIdx.x;
    if (idx >= total) return;
    const int e = idx / chunks;
    const int c = (idx - e * chunks) << 2;
    const int s = src[e], d = dst[e];
    const float w = an[e];
    const float a = wf * w, b = wb * w;
    const float* xs = x + (size_t)s * ldx + c;
    const float* xd = x + (size_t)d * ldx + c;
    float* od = out + (size_t)d * ldo + c;
    float* os = out + (size_t)s * ldo + c;
    #pragma unroll
    for (int j = 0; j < 4; ++j) {
        atomicAdd(od + j, a * xs[j]);
        atomicAdd(os + j, b * xd[j]);
    }
}

// logits = [r,i] @ Wc^T + bc; out = log_softmax(logits)
__global__ void k_classify(const float* __restrict__ r, const float* __restrict__ im,
                           const float* __restrict__ Wc, const float* __restrict__ bc,
                           float* __restrict__ out, int N)
{
    int n = blockIdx.x * blockDim.x + threadIdx.x;
    if (n >= N) return;
    const float* rn  = r  + (size_t)n * 64;
    const float* in_ = im + (size_t)n * 64;
    float logit[10];
    #pragma unroll
    for (int c = 0; c < 10; ++c) {
        const float* w = Wc + c * 128;
        float s = bc[c];
        for (int f = 0; f < 64; ++f) s += w[f]      * rn[f];
        for (int f = 0; f < 64; ++f) s += w[64 + f] * in_[f];
        logit[c] = s;
    }
    float m = logit[0];
    #pragma unroll
    for (int c = 1; c < 10; ++c) m = fmaxf(m, logit[c]);
    float sum = 0.0f;
    #pragma unroll
    for (int c = 0; c < 10; ++c) sum += expf(logit[c] - m);
    const float ls = logf(sum);
    #pragma unroll
    for (int c = 0; c < 10; ++c) out[(size_t)n * 10 + c] = logit[c] - m - ls;
}

// ---------------------------------------------------------------------------
// Host orchestration
// ---------------------------------------------------------------------------
extern "C" void kernel_launch(void* const* d_in, const int* in_sizes, int n_in,
                              void* d_out, int out_size, void* d_ws, size_t ws_size,
                              hipStream_t stream)
{
    (void)n_in; (void)out_size; (void)ws_size;

    const float* xr = (const float*)d_in[0];
    const float* xi = (const float*)d_in[1];
    const int*   ei = (const int*)  d_in[2];
    const float* W1 = (const float*)d_in[3];   // [3,128,64] == [384,64]
    const float* b1 = (const float*)d_in[4];
    const float* W2 = (const float*)d_in[5];   // [3,64,64]  == [192,64]
    const float* b2 = (const float*)d_in[6];
    const float* Wc = (const float*)d_in[7];
    const float* bc = (const float*)d_in[8];

    const int N = in_sizes[0] / 128;
    const int E = in_sizes[2] / 2;
    const int* src = ei;
    const int* dst = ei + E;

    // Laplacian constants (match reference float math)
    const float th     = 6.2831853071795864769f * 0.25f;   // 2*pi*q
    const float cv     = cosf(th);                          // ~ -4.4e-8 in f32
    const float sv     = sinf(th);                          // 1.0
    const float scale  = 2.0f / 2.0f;                       // 2/lambda_max
    const float diag_r = scale * 1.0f - 1.0f;               // 0
    const float diag_i = -1.0f;

    // workspace layout (floats)
    float* p = (float*)d_ws;
    float* deg  = p;  p += N;
    float* dinv = p;  p += N;
    float* an   = p;  p += E;
    float* Aa   = p;  p += (size_t)N * 384;   // concatenated-K chain A [N,3F]
    float* Ab   = p;  p += (size_t)N * 384;   // concatenated-K chain B
    float* hr   = p;  p += (size_t)N * 64;    // conv1 outputs
    float* hi   = p;  p += (size_t)N * 64;
    float* fr   = p;  p += (size_t)N * 64;    // conv2 outputs
    float* fi   = p;  p += (size_t)N * 64;

    const int TB = 256;
    auto blk = [&](int n) { return dim3((n + TB - 1) / TB); };

    auto gemm2 = [&](const float* A0, const float* B0, const float* Wk, const float* bias,
                     float* oR, float* oI, int kdim) {
        gemm2_wmma_relu<<<dim3((N + 127) / 128), dim3(256), 0, stream>>>(
            A0, B0, Wk, bias, oR, oI, N, kdim);
    };
    auto scatter = [&](const float* x, int ldx, float* o, int ldo, int F, float wf, float wb) {
        int total = E * (F >> 2);
        k_scatter<<<blk(total), dim3(TB), 0, stream>>>(src, dst, an, x, ldx, o, ldo, E, F, wf, wb);
    };
    auto axpby = [&](float* o, int ldo, const float* x, int ldx, float al,
                     const float* y, int ldy, float be, int F) {
        k_axpby<<<blk(N * F), dim3(TB), 0, stream>>>(o, ldo, x, ldx, al, y, ldy, be, N, F);
    };

    // ---- magnetic Laplacian edge weights ----
    k_fill<<<blk(N), dim3(TB), 0, stream>>>(deg, 0.0f, N);
    k_deg<<<blk(E), dim3(TB), 0, stream>>>(src, dst, deg, E);
    k_dinv<<<blk(N), dim3(TB), 0, stream>>>(deg, dinv, N);
    k_anorm<<<blk(E), dim3(TB), 0, stream>>>(src, dst, dinv, an, E);

    // ================= conv1: F = 128 -> 64 =================
    {
        const int F = 128, LD = 384;
        float* T0a = Aa;           float* T0b = Ab;
        float* T1a = Aa + F;       float* T1b = Ab + F;
        float* T2a = Aa + 2 * F;   float* T2b = Ab + 2 * F;

        // T0 = x
        axpby(T0a, LD, xr, F, 1.0f, xr, F, 0.0f, F);
        axpby(T0b, LD, xi, F, 1.0f, xi, F, 0.0f, F);
        // T1 = prop(x, ew, diag)
        axpby(T1a, LD, xr, F, diag_r, xr, F, 0.0f, F);
        scatter(xr, F, T1a, LD, F, -cv, -cv);          // ew_r (both dirs)
        axpby(T1b, LD, xi, F, diag_i, xi, F, 0.0f, F);
        scatter(xi, F, T1b, LD, F, -sv, +sv);          // ew_i (theta sign flips)
        // T2 = 2*prop(T1, ew_r, diag_r) - T0   (source quirk: ew_r for all k>=2)
        axpby(T2a, LD, T1a, LD, 2.0f * diag_r, xr, F, -1.0f, F);
        scatter(T1a, LD, T2a, LD, F, -2.0f * cv, -2.0f * cv);
        axpby(T2b, LD, T1b, LD, 2.0f * diag_r, xi, F, -1.0f, F);
        scatter(T1b, LD, T2b, LD, F, -2.0f * cv, -2.0f * cv);
        // fused dual GEMM + combine + complex ReLU
        gemm2(Aa, Ab, W1, b1, hr, hi, 3 * F);
    }

    // ================= conv2: F = 64 -> 64 =================
    {
        const int F = 64, LD = 192;
        float* T0a = Aa;           float* T0b = Ab;
        float* T1a = Aa + F;       float* T1b = Ab + F;
        float* T2a = Aa + 2 * F;   float* T2b = Ab + 2 * F;

        axpby(T0a, LD, hr, F, 1.0f, hr, F, 0.0f, F);
        axpby(T0b, LD, hi, F, 1.0f, hi, F, 0.0f, F);
        axpby(T1a, LD, hr, F, diag_r, hr, F, 0.0f, F);
        scatter(hr, F, T1a, LD, F, -cv, -cv);
        axpby(T1b, LD, hi, F, diag_i, hi, F, 0.0f, F);
        scatter(hi, F, T1b, LD, F, -sv, +sv);
        axpby(T2a, LD, T1a, LD, 2.0f * diag_r, hr, F, -1.0f, F);
        scatter(T1a, LD, T2a, LD, F, -2.0f * cv, -2.0f * cv);
        axpby(T2b, LD, T1b, LD, 2.0f * diag_r, hi, F, -1.0f, F);
        scatter(T1b, LD, T2b, LD, F, -2.0f * cv, -2.0f * cv);
        gemm2(Aa, Ab, W2, b2, fr, fi, 3 * F);
    }

    // ---- classifier + log_softmax ----
    k_classify<<<blk(N), dim3(TB), 0, stream>>>(fr, fi, Wc, bc, (float*)d_out, N);
}
